// HybridHopfieldLayer_21835613733000
// MI455X (gfx1250) — matrix operand order repaired and checked by reading, not compiled, not run on hardware
//
#include <hip/hip_runtime.h>

#define B_ 4
#define S_ 2048
#define D_ 1024
#define H_ 1024

typedef __attribute__((ext_vector_type(16))) __bf16 bf16x16;
typedef __attribute__((ext_vector_type(8)))  float  f32x8;
typedef __attribute__((ext_vector_type(8)))  unsigned short u16x8;

// Type matching the async-to-LDS builtin's parameter: GCC-style int4 vector.
typedef int v4i __attribute__((vector_size(4 * sizeof(int))));
typedef __attribute__((address_space(1))) v4i* gv4i_ptr;
typedef __attribute__((address_space(3))) v4i* lv4i_ptr;

union Frag { bf16x16 v; u16x8 h[2]; };

#if __has_builtin(__builtin_amdgcn_global_load_async_to_lds_b128)
#define HAS_ASYNC 1
#else
#define HAS_ASYNC 0
#endif

#if HAS_ASYNC
#if __has_builtin(__builtin_amdgcn_s_wait_asynccnt)
#define WAIT_ASYNC() __builtin_amdgcn_s_wait_asynccnt(0)
#else
#define WAIT_ASYNC() asm volatile("s_wait_asynccnt 0x0" ::: "memory")
#endif
#else
#define WAIT_ASYNC()
#endif

__device__ __forceinline__ unsigned short f2bf(float f) {
  unsigned int u = __float_as_uint(f);
  u += 0x7FFFu + ((u >> 16) & 1u);        // round-to-nearest-even
  return (unsigned short)(u >> 16);
}

__device__ __forceinline__ f32x8 wmma_bf16(const Frag& a, const Frag& b, f32x8 c) {
  return __builtin_amdgcn_wmma_f32_16x16x32_bf16(false, a.v, false, b.v,
                                                 (short)0, c, false, false);
}

// ---------------- sparsity = mean(|x|) ----------------
__global__ void zero_kernel(float* p) {
  if (threadIdx.x == 0 && blockIdx.x == 0) *p = 0.f;
}

__global__ __launch_bounds__(256) void absmean_kernel(const float* __restrict__ x, int n,
                                                      float* __restrict__ out) {
  float s = 0.f;
  for (int i = blockIdx.x * blockDim.x + threadIdx.x; i < n; i += gridDim.x * blockDim.x)
    s += fabsf(x[i]);
  __shared__ float red[256];
  red[threadIdx.x] = s; __syncthreads();
  for (int t = 128; t > 0; t >>= 1) {
    if ((int)threadIdx.x < t) red[threadIdx.x] += red[threadIdx.x + t];
    __syncthreads();
  }
  if (threadIdx.x == 0) atomicAdd(out, red[0] / (float)n);
}

// ---------------- fp32 -> bf16 ----------------
__global__ __launch_bounds__(256) void f2bf_kernel(const float* __restrict__ in,
                                                   unsigned short* __restrict__ out, int n) {
  for (int i = blockIdx.x * blockDim.x + threadIdx.x; i < n; i += gridDim.x * blockDim.x)
    out[i] = f2bf(in[i]);
}

// ---------------- cooperative GEMM core ----------------
// Block = 256 threads (8 waves), block tile 128x128, K-step 32, double-buffered LDS,
// async global->LDS staging. A and B are both row-major [rows][ld] with K contiguous
// (B is the [N][K] "weights-like" matrix). Caller pre-offsets Ag/Bg to the block tile.

#define LDSPAD 40   // 32 data + 8 pad shorts; 80B rows: 16B aligned, conflict-free

__device__ __forceinline__ void stage_tile(const unsigned short* __restrict__ g, size_t ld,
                                           int kk, unsigned short (&dst)[128][LDSPAD], int tid) {
#pragma unroll
  for (int c = 0; c < 2; ++c) {
    int chunk = tid * 2 + c;            // 512 chunks of 16B cover 128x32 bf16
    int row = chunk >> 2;
    int col = (chunk & 3) * 8;
    const unsigned short* src = g + (size_t)row * ld + kk + col;
    unsigned short* d = &dst[row][col];
#if HAS_ASYNC
    __builtin_amdgcn_global_load_async_to_lds_b128((gv4i_ptr)(src), (lv4i_ptr)(d), 0, 0);
#else
    *(u16x8*)d = *(const u16x8*)src;
#endif
  }
}

__device__ __forceinline__ void gemm_core(const unsigned short* __restrict__ Ag, size_t lda,
                                          const unsigned short* __restrict__ Bg, size_t ldb,
                                          int K, f32x8 (&acc)[8]) {
  __shared__ unsigned short As[2][128][LDSPAD];
  __shared__ unsigned short Bs[2][128][LDSPAD];
  const int tid = threadIdx.x;
  const int lane = tid & 31, wave = tid >> 5;
  const int waveM = (wave & 3) * 32;    // 4 M sub-blocks of 32
  const int waveN = (wave >> 2) * 64;   // 2 N sub-blocks of 64
  const int half = lane >> 4, l15 = lane & 15;
  const int ka = half * 8;              // A frag K base (ISA 16-bit A layout)
  const int kb = half * 8;              // B frag K base (ISA 16-bit layout, same as A)

#pragma unroll
  for (int t = 0; t < 8; ++t)
#pragma unroll
    for (int r = 0; r < 8; ++r) acc[t][r] = 0.f;

  stage_tile(Ag, lda, 0, As[0], tid);
  stage_tile(Bg, ldb, 0, Bs[0], tid);
  WAIT_ASYNC();
  __syncthreads();

  for (int kk = 0; kk < K; kk += 32) {
    const int cur = (kk >> 5) & 1;
    const bool more = (kk + 32) < K;
    if (more) {
      stage_tile(Ag, lda, kk + 32, As[cur ^ 1], tid);
      stage_tile(Bg, ldb, kk + 32, Bs[cur ^ 1], tid);
    }

    Frag a[2], b[4];
#pragma unroll
    for (int mt = 0; mt < 2; ++mt) {
      const unsigned short* p = &As[cur][waveM + mt * 16 + l15][ka];
      a[mt].h[0] = *(const u16x8*)(p);
      a[mt].h[1] = *(const u16x8*)(p + 16);
    }
#pragma unroll
    for (int nt = 0; nt < 4; ++nt) {
      const unsigned short* p = &Bs[cur][waveN + nt * 16 + l15][kb];
      b[nt].h[0] = *(const u16x8*)(p);
      b[nt].h[1] = *(const u16x8*)(p + 16);
    }
#pragma unroll
    for (int mt = 0; mt < 2; ++mt)
#pragma unroll
      for (int nt = 0; nt < 4; ++nt)
        acc[mt * 4 + nt] = wmma_bf16(a[mt], b[nt], acc[mt * 4 + nt]);

    if (more) WAIT_ASYNC();
    __syncthreads();
  }
}

// ---------------- QKV projection: [B*S,D] x W[H,D]^T ----------------
__global__ __launch_bounds__(256) void gemm_qkv(const unsigned short* __restrict__ xb,
                                                const unsigned short* __restrict__ wb,
                                                unsigned short* __restrict__ Q,
                                                unsigned short* __restrict__ K,
                                                unsigned short* __restrict__ Vt) {
  const int m0 = blockIdx.x * 128;
  const int n0 = blockIdx.y * 128;
  const int which = blockIdx.z;
  const unsigned short* W = wb + (size_t)which * H_ * D_;

  f32x8 acc[8];
  gemm_core(xb + (size_t)m0 * D_, D_, W + (size_t)n0 * D_, D_, D_, acc);

  const int tid = threadIdx.x, lane = tid & 31, wave = tid >> 5;
  const int waveM = (wave & 3) * 32, waveN = (wave >> 2) * 64;
  const int half = lane >> 4, l15 = lane & 15;
#pragma unroll
  for (int mt = 0; mt < 2; ++mt)
#pragma unroll
    for (int nt = 0; nt < 4; ++nt)
#pragma unroll
      for (int r = 0; r < 8; ++r) {
        int m = m0 + waveM + mt * 16 + half * 8 + r;
        int n = n0 + waveN + nt * 16 + l15;
        unsigned short v = f2bf(acc[mt * 4 + nt][r]);
        if (which == 0)      Q[(size_t)m * H_ + n] = v;
        else if (which == 1) K[(size_t)m * H_ + n] = v;
        else {               // V stored transposed per batch: Vt[b][h][t]
          int b = m >> 11;           // / S_
          int s = m & (S_ - 1);
          Vt[((size_t)b * H_ + n) * S_ + s] = v;
        }
      }
}

// ---------------- energy = Q K^T * (1/sqrt(H)) ----------------
__global__ __launch_bounds__(256) void gemm_energy(const unsigned short* __restrict__ Q,
                                                   const unsigned short* __restrict__ K,
                                                   float* __restrict__ scores) {
  const int m0 = blockIdx.x * 128;     // s
  const int n0 = blockIdx.y * 128;     // t
  const int b  = blockIdx.z;
  const unsigned short* Qb = Q + (size_t)b * S_ * H_;
  const unsigned short* Kb = K + (size_t)b * S_ * H_;

  f32x8 acc[8];
  gemm_core(Qb + (size_t)m0 * H_, H_, Kb + (size_t)n0 * H_, H_, H_, acc);

  const float scale = 0.03125f;        // 1/sqrt(1024) * BETA
  float* out = scores + (size_t)b * S_ * S_;
  const int tid = threadIdx.x, lane = tid & 31, wave = tid >> 5;
  const int waveM = (wave & 3) * 32, waveN = (wave >> 2) * 64;
  const int half = lane >> 4, l15 = lane & 15;
#pragma unroll
  for (int mt = 0; mt < 2; ++mt)
#pragma unroll
    for (int nt = 0; nt < 4; ++nt)
#pragma unroll
      for (int r = 0; r < 8; ++r) {
        int m = m0 + waveM + mt * 16 + half * 8 + r;
        int n = n0 + waveN + nt * 16 + l15;
        out[(size_t)m * S_ + n] = acc[mt * 4 + nt][r] * scale;
      }
}

// ---------------- out = attention x V (via Vt) ----------------
__global__ __launch_bounds__(256) void gemm_av(const unsigned short* __restrict__ att,
                                               const unsigned short* __restrict__ Vt,
                                               float* __restrict__ out) {
  const int m0 = blockIdx.x * 128;     // s
  const int n0 = blockIdx.y * 128;     // h
  const int b  = blockIdx.z;
  const unsigned short* Ab = att + (size_t)b * S_ * S_;
  const unsigned short* Vb = Vt + (size_t)b * H_ * S_;

  f32x8 acc[8];
  gemm_core(Ab + (size_t)m0 * S_, S_, Vb + (size_t)n0 * S_, S_, S_, acc);

  float* ob = out + (size_t)b * S_ * H_;
  const int tid = threadIdx.x, lane = tid & 31, wave = tid >> 5;
  const int waveM = (wave & 3) * 32, waveN = (wave >> 2) * 64;
  const int half = lane >> 4, l15 = lane & 15;
#pragma unroll
  for (int mt = 0; mt < 2; ++mt)
#pragma unroll
    for (int nt = 0; nt < 4; ++nt)
#pragma unroll
      for (int r = 0; r < 8; ++r) {
        int m = m0 + waveM + mt * 16 + half * 8 + r;
        int n = n0 + waveN + nt * 16 + l15;
        ob[(size_t)m * H_ + n] = acc[mt * 4 + nt][r];
      }
}

// ---------------- per-row sparsemax (or softmax), row length 2048 ----------------
__global__ __launch_bounds__(256) void attn_rows(const float* __restrict__ scores,
                                                 unsigned short* __restrict__ att,
                                                 const float* __restrict__ sparsity) {
  const int row = blockIdx.x;
  const int tid = threadIdx.x;
  const int N = S_, NT = 256;
  __shared__ float z[S_];
  __shared__ float sz[S_];
  __shared__ float red[256];

  const float* rs = scores + (size_t)row * N;
  unsigned short* ra = att + (size_t)row * N;

  float lv[8];
  float lmax = -3.402823466e38f;
#pragma unroll
  for (int e = 0; e < 8; ++e) {
    lv[e] = rs[tid + e * NT];
    lmax = fmaxf(lmax, lv[e]);
  }
  red[tid] = lmax; __syncthreads();
  for (int s = NT / 2; s > 0; s >>= 1) { if (tid < s) red[tid] = fmaxf(red[tid], red[tid + s]); __syncthreads(); }
  float rmax = red[0]; __syncthreads();

  float lsum = 0.f;
#pragma unroll
  for (int e = 0; e < 8; ++e) {
    float v = lv[e] - rmax;
    z[tid + e * NT] = v;
    lsum += v;
  }
  red[tid] = lsum; __syncthreads();
  for (int s = NT / 2; s > 0; s >>= 1) { if (tid < s) red[tid] += red[tid + s]; __syncthreads(); }
  float zsum = red[0]; __syncthreads();

  if (*sparsity < 1.0f) {
    // ---- sparsemax (faithful, per-position tau) ----
    float step = (zsum - 1.0f) / (float)N;
#pragma unroll
    for (int e = 0; e < 8; ++e) sz[tid + e * NT] = z[tid + e * NT];
    __syncthreads();
    // bitonic sort, descending
    for (unsigned k = 2; k <= (unsigned)N; k <<= 1) {
      for (unsigned j = k >> 1; j > 0; j >>= 1) {
        for (unsigned i = (unsigned)tid; i < (unsigned)N; i += NT) {
          unsigned ixj = i ^ j;
          if (ixj > i) {
            float a = sz[i], b = sz[ixj];
            bool swp = ((i & k) == 0) ? (a < b) : (a > b);
            if (swp) { sz[i] = b; sz[ixj] = a; }
          }
        }
        __syncthreads();
      }
    }
    // k = max((i+1) * [sorted_i > step])
    float lk = 0.f;
#pragma unroll
    for (int e = 0; e < 8; ++e) {
      int i = tid + e * NT;
      if (sz[i] > step) lk = fmaxf(lk, (float)(i + 1));
    }
    red[tid] = lk; __syncthreads();
    for (int s = NT / 2; s > 0; s >>= 1) { if (tid < s) red[tid] = fmaxf(red[tid], red[tid + s]); __syncthreads(); }
    float kf = red[0]; __syncthreads();
    // inclusive cumsum of sorted values in place (blocked scan)
    float run = 0.f;
    int base = tid * 8;
#pragma unroll
    for (int e = 0; e < 8; ++e) { run += sz[base + e]; sz[base + e] = run; }
    red[tid] = run; __syncthreads();
    for (int off = 1; off < NT; off <<= 1) {
      float v = red[tid];
      float add = (tid >= off) ? red[tid - off] : 0.f;
      __syncthreads();
      red[tid] = v + add;
      __syncthreads();
    }
    float ofs = (tid > 0) ? red[tid - 1] : 0.f;
#pragma unroll
    for (int e = 0; e < 8; ++e) sz[base + e] += ofs;
    __syncthreads();
    float inv_k = 1.0f / kf;
#pragma unroll
    for (int e = 0; e < 8; ++e) {
      int i = tid + e * NT;
      float tau = (sz[i] - 1.0f - step * kf) * inv_k;
      float o = z[i] - tau;
      ra[i] = f2bf(o > 0.f ? o : 0.f);
    }
  } else {
    // ---- softmax fallback ----
    float ev[8];
    float ls = 0.f;
#pragma unroll
    for (int e = 0; e < 8; ++e) { ev[e] = __expf(z[tid + e * NT]); ls += ev[e]; }
    red[tid] = ls; __syncthreads();
    for (int s = NT / 2; s > 0; s >>= 1) { if (tid < s) red[tid] += red[tid + s]; __syncthreads(); }
    float inv = 1.0f / red[0];
#pragma unroll
    for (int e = 0; e < 8; ++e) ra[tid + e * NT] = f2bf(ev[e] * inv);
  }
}

extern "C" void kernel_launch(void* const* d_in, const int* in_sizes, int n_in,
                              void* d_out, int out_size, void* d_ws, size_t ws_size,
                              hipStream_t stream) {
  (void)in_sizes; (void)n_in; (void)out_size; (void)ws_size;
  const float* x  = (const float*)d_in[0];
  const float* Wq = (const float*)d_in[1];
  const float* Wk = (const float*)d_in[2];
  const float* Wv = (const float*)d_in[3];
  float* out = (float*)d_out;

  char* ws = (char*)d_ws;
  const size_t XN = (size_t)B_ * S_ * D_;   // 8,388,608 (== B*S*H)
  const size_t WN = (size_t)H_ * D_;        // 1,048,576
  size_t off = 0;
  float* sp            = (float*)(ws + off);          off += 256;
  unsigned short* xb   = (unsigned short*)(ws + off); off += XN * 2;
  unsigned short* wb   = (unsigned short*)(ws + off); off += 3 * WN * 2;
  unsigned short* Qb   = (unsigned short*)(ws + off); off += XN * 2;
  unsigned short* Kb   = (unsigned short*)(ws + off); off += XN * 2;
  unsigned short* Vt   = (unsigned short*)(ws + off); off += XN * 2;
  float* sc            = (float*)(ws + off);          off += (size_t)B_ * S_ * S_ * 4;
  unsigned short* att  = (unsigned short*)(ws + off); off += (size_t)B_ * S_ * S_ * 2;

  zero_kernel<<<1, 1, 0, stream>>>(sp);
  absmean_kernel<<<512, 256, 0, stream>>>(x, (int)XN, sp);

  f2bf_kernel<<<4096, 256, 0, stream>>>(x,  xb, (int)XN);
  f2bf_kernel<<<1024, 256, 0, stream>>>(Wq, wb,           (int)WN);
  f2bf_kernel<<<1024, 256, 0, stream>>>(Wk, wb + WN,      (int)WN);
  f2bf_kernel<<<1024, 256, 0, stream>>>(Wv, wb + 2 * WN,  (int)WN);

  gemm_qkv<<<dim3((B_ * S_) / 128, H_ / 128, 3), 256, 0, stream>>>(xb, wb, Qb, Kb, Vt);
  gemm_energy<<<dim3(S_ / 128, S_ / 128, B_), 256, 0, stream>>>(Qb, Kb, sc);
  attn_rows<<<B_ * S_, 256, 0, stream>>>(sc, att, sp);
  gemm_av<<<dim3(S_ / 128, H_ / 128, B_), 256, 0, stream>>>(att, Vt, out);
}
